// MuiltHead_SelfAttention_2551210574512
// MI455X (gfx1250) — matrix operand order repaired and checked by reading, not compiled, not run on hardware
//
#include <hip/hip_runtime.h>
#include <hip/hip_bf16.h>

// ---------------------------------------------------------------------------
// Multi-head self-attention for gfx1250 (CDNA5, wave32, WMMA).
// B=8, N=1024, H=12, Dh=64, D_IN=D_OUT=768, 3*H*Dh=2304.
// ~64 GFLOP total vs ~100 MB of traffic -> compute bound; everything runs on
// v_wmma_f32_16x16x32_f16 (f16 operands, f32 accumulate).
// ---------------------------------------------------------------------------

typedef __attribute__((ext_vector_type(16))) _Float16 v16h;
typedef __attribute__((ext_vector_type(8)))  float    v8f;
typedef int async_i32x4 __attribute__((vector_size(4 * sizeof(int))));

#define CB 8
#define CN 1024
#define CH 12
#define CDH 64
#define CDIN 768
#define CDQKV 2304
#define CDOUT 768

#if defined(__AMDGCN__) && __has_builtin(__builtin_amdgcn_global_load_async_to_lds_b128)
#define HAVE_ASYNC_COPY 1
#else
#define HAVE_ASYNC_COPY 0
#endif

__device__ __forceinline__ void wait_async_zero() {
#if HAVE_ASYNC_COPY
#if __has_builtin(__builtin_amdgcn_s_wait_asynccnt)
    __builtin_amdgcn_s_wait_asynccnt(0);
#else
    asm volatile("s_wait_asynccnt 0x0" ::: "memory");
#endif
#endif
}

// Load a 16-half WMMA fragment as two 16-byte LDS/global reads.
__device__ __forceinline__ v16h load_frag16(const _Float16* p0, const _Float16* p1) {
    v16h r;
    ((uint4*)&r)[0] = *(const uint4*)p0;
    ((uint4*)&r)[1] = *(const uint4*)p1;
    return r;
}

// ---------------------------------------------------------------------------
// GEMM staging helpers: global -> registers, registers -> LDS (f16).
// MODE 0: A is f32. MODE 1: A is f16.
// ---------------------------------------------------------------------------
template <int MODE>
__device__ __forceinline__ void gemm_gload(
    const void* __restrict__ Aptr, const float* __restrict__ Wptr,
    int K, int Nc, int mBase, int nBase, int k0, int tid,
    float4 (&aF)[4], uint4 (&aH)[2], float4 (&bF)[4])
{
    if (MODE == 0) {
        const float* A = (const float*)Aptr;
        #pragma unroll
        for (int r = 0; r < 4; ++r) {
            int idx = tid + r * 256;          // 1024 float4 chunks: 128 rows x 8
            int row = idx >> 3;
            int c4  = (idx & 7) * 4;
            aF[r] = *(const float4*)&A[(size_t)(mBase + row) * K + k0 + c4];
        }
    } else {
        const _Float16* A = (const _Float16*)Aptr;
        #pragma unroll
        for (int r = 0; r < 2; ++r) {
            int idx = tid + r * 256;          // 512 chunks of 8 halfs
            int row = idx >> 2;
            int c8  = (idx & 3) * 8;
            aH[r] = *(const uint4*)&A[(size_t)(mBase + row) * K + k0 + c8];
        }
    }
    #pragma unroll
    for (int r = 0; r < 4; ++r) {
        int idx = tid + r * 256;              // 1024 float4 chunks: 32 k-rows x 32
        int kr  = idx >> 5;
        int c4  = (idx & 31) * 4;
        bF[r] = *(const float4*)&Wptr[(size_t)(k0 + kr) * Nc + nBase + c4];
    }
}

template <int MODE>
__device__ __forceinline__ void gemm_lstore(
    _Float16* __restrict__ As, _Float16* __restrict__ Bt, int tid,
    const float4 (&aF)[4], const uint4 (&aH)[2], const float4 (&bF)[4])
{
    if (MODE == 0) {
        #pragma unroll
        for (int r = 0; r < 4; ++r) {
            int idx = tid + r * 256;
            int row = idx >> 3;
            int c4  = (idx & 7) * 4;
            _Float16* dst = &As[row * 32 + c4];
            dst[0] = (_Float16)aF[r].x; dst[1] = (_Float16)aF[r].y;
            dst[2] = (_Float16)aF[r].z; dst[3] = (_Float16)aF[r].w;
        }
    } else {
        #pragma unroll
        for (int r = 0; r < 2; ++r) {
            int idx = tid + r * 256;
            int row = idx >> 2;
            int c8  = (idx & 3) * 8;
            *(uint4*)&As[row * 32 + c8] = aH[r];
        }
    }
    #pragma unroll
    for (int r = 0; r < 4; ++r) {              // W stored transposed: Bt[n][k]
        int idx = tid + r * 256;
        int kr  = idx >> 5;
        int c4  = (idx & 31) * 4;
        Bt[(c4 + 0) * 32 + kr] = (_Float16)bF[r].x;
        Bt[(c4 + 1) * 32 + kr] = (_Float16)bF[r].y;
        Bt[(c4 + 2) * 32 + kr] = (_Float16)bF[r].z;
        Bt[(c4 + 3) * 32 + kr] = (_Float16)bF[r].w;
    }
}

// ---------------------------------------------------------------------------
// Tiled GEMM: C = A @ W + bias.  Block tile 128x128, K-step 32, 8 waves,
// double-buffered LDS (one barrier per K-step, global loads overlap WMMA).
// Each wave computes a 64x32 sub-tile (4x2 WMMA accumulators).
// MODE 0: epilogue scatters f16 into q/k/v [B,H,N,Dh] workspace.
// MODE 1: epilogue stores f32 with bias.
// ---------------------------------------------------------------------------
template <int MODE>
__global__ __launch_bounds__(256) void mhsa_gemm_kernel(
    const void* __restrict__ Aptr, const float* __restrict__ Wptr,
    const float* __restrict__ bias, void* __restrict__ Cptr,
    int M, int K, int Nc)
{
    __shared__ __align__(16) _Float16 As[2][128 * 32];
    __shared__ __align__(16) _Float16 Bt[2][128 * 32];

    const int tid  = threadIdx.x;
    const int wave = tid >> 5;
    const int lane = tid & 31;
    const int wm   = (wave >> 2) * 64;
    const int wn   = (wave & 3) * 32;
    const int mBase = blockIdx.y * 128;
    const int nBase = blockIdx.x * 128;

    v8f acc[4][2] = {};

    const int arow = lane & 15;
    const int akb  = (lane >> 4) * 8;    // A-frag K group per ISA layout
    const int bcol = lane & 15;
    const int bkb  = (lane >> 4) * 8;    // B-frag K offset per ISA layout

    float4 aF[4]; uint4 aH[2]; float4 bF[4];
    const int nsteps = K / 32;

    // prologue: stage step 0 into buffer 0
    gemm_gload<MODE>(Aptr, Wptr, K, Nc, mBase, nBase, 0, tid, aF, aH, bF);
    gemm_lstore<MODE>(As[0], Bt[0], tid, aF, aH, bF);
    __syncthreads();

    int buf = 0;
    for (int it = 0; it < nsteps; ++it) {
        const bool hasNext = (it + 1 < nsteps);
        if (hasNext)
            gemm_gload<MODE>(Aptr, Wptr, K, Nc, mBase, nBase, (it + 1) * 32, tid, aF, aH, bF);

        const _Float16* asb = As[buf];
        const _Float16* btb = Bt[buf];
        v16h afrag[4], bfrag[2];
        #pragma unroll
        for (int ms = 0; ms < 4; ++ms) {
            const _Float16* p = &asb[(wm + ms * 16 + arow) * 32];
            afrag[ms] = load_frag16(p + akb, p + 16 + akb);
        }
        #pragma unroll
        for (int ns = 0; ns < 2; ++ns) {
            const _Float16* p = &btb[(wn + ns * 16 + bcol) * 32 + bkb];
            bfrag[ns] = load_frag16(p, p + 16);
        }
        #pragma unroll
        for (int ms = 0; ms < 4; ++ms)
            #pragma unroll
            for (int ns = 0; ns < 2; ++ns)
                acc[ms][ns] = __builtin_amdgcn_wmma_f32_16x16x32_f16(
                    false, afrag[ms], false, bfrag[ns],
                    (short)0, acc[ms][ns], false, false);

        if (hasNext)
            gemm_lstore<MODE>(As[buf ^ 1], Bt[buf ^ 1], tid, aF, aH, bF);
        __syncthreads();
        buf ^= 1;
    }

    // ---- epilogue ----
    const int rowhi = (lane >> 4) * 8;
    const int colin = lane & 15;
    #pragma unroll
    for (int ms = 0; ms < 4; ++ms) {
        #pragma unroll
        for (int ns = 0; ns < 2; ++ns) {
            #pragma unroll
            for (int r = 0; r < 8; ++r) {
                int row = mBase + wm + ms * 16 + rowhi + r;
                int col = nBase + wn + ns * 16 + colin;
                float vv = acc[ms][ns][r] + bias[col];
                if (MODE == 0) {
                    _Float16* qkv = (_Float16*)Cptr;
                    int sel = col / (CH * CDH);
                    int hh  = (col % (CH * CDH)) / CDH;
                    int dd  = col & (CDH - 1);
                    int b   = row >> 10;
                    int n   = row & (CN - 1);
                    size_t off = (size_t)sel * ((size_t)CB * CH * CN * CDH)
                               + ((size_t)(b * CH + hh) * CN + n) * CDH + dd;
                    qkv[off] = (_Float16)vv;
                } else {
                    ((float*)Cptr)[(size_t)row * Nc + col] = vv;
                }
            }
        }
    }
}

// ---------------------------------------------------------------------------
// Flash attention with swapped roles: S = K·Qᵀ (softmax over q index), Y = P·V.
// One block = 8 waves = 128 output rows of one (b,h). j-loop in steps of 64:
// 16 WMMA per iteration (8 for S, 8 for P·V), 2 barriers per iteration.
// Q tile staged with async global->LDS copies (ASYNCcnt-tracked) where the
// toolchain exposes them.
// ---------------------------------------------------------------------------
__global__ __launch_bounds__(256) void mhsa_attn_kernel(
    const _Float16* __restrict__ q, const _Float16* __restrict__ k,
    const _Float16* __restrict__ v, _Float16* __restrict__ y)
{
    __shared__ __align__(16) _Float16 Qs[64 * 64];        // [j][d]
    __shared__ __align__(16) _Float16 Vt[64 * 64];        // [d][j] (transposed)
    __shared__ __align__(16) _Float16 Ps[8 * 16 * 64];    // per-wave P tile (C->A relayout)

    const int tid  = threadIdx.x;
    const int wave = tid >> 5;
    const int lane = tid & 31;
    const int bh   = blockIdx.x >> 3;            // N/128 = 8 row-blocks per (b,h)
    const int iBlk = blockIdx.x & 7;
    const int iBase = iBlk * 128 + wave * 16;

    const _Float16* qp = q + (size_t)bh * CN * CDH;
    const _Float16* kp = k + (size_t)bh * CN * CDH;
    const _Float16* vp = v + (size_t)bh * CN * CDH;

    const int l15  = lane & 15;
    const int akb  = (lane >> 4) * 8;
    const int bkoff = (lane >> 4) * 8;

    // A fragments of the 16 "query" rows (= k rows), d = 0..63, loop-invariant.
    const _Float16* krow = kp + (size_t)(iBase + l15) * CDH;
    const v16h aK0 = load_frag16(krow + akb,      krow + 16 + akb);
    const v16h aK1 = load_frag16(krow + 32 + akb, krow + 48 + akb);

    v8f Yacc[4] = {};
    float mrow[8], lrow[8];
    #pragma unroll
    for (int r = 0; r < 8; ++r) { mrow[r] = -1e30f; lrow[r] = 0.f; }

    const float scale = 0.125f;  // 1/sqrt(64)

    for (int jb = 0; jb < CN; jb += 64) {
        // ---- stage Q block [64][64] and V block transposed [64 d][64 j] ----
        #pragma unroll
        for (int rr = 0; rr < 2; ++rr) {
            int idx = tid + rr * 256;             // 512 chunks of 8 halfs
            int j   = idx >> 3;
            int d8  = (idx & 7) * 8;
#if HAVE_ASYNC_COPY
            __builtin_amdgcn_global_load_async_to_lds_b128(
                (__attribute__((address_space(1))) async_i32x4*)(qp + (size_t)(jb + j) * CDH + d8),
                (__attribute__((address_space(3))) async_i32x4*)&Qs[j * 64 + d8],
                0, 0);
#else
            *(uint4*)&Qs[j * 64 + d8] = *(const uint4*)&qp[(size_t)(jb + j) * CDH + d8];
#endif
            uint4 raw = *(const uint4*)&vp[(size_t)(jb + j) * CDH + d8];
            const _Float16* hh = (const _Float16*)&raw;
            #pragma unroll
            for (int e = 0; e < 8; ++e) Vt[(d8 + e) * 64 + j] = hh[e];
            if (jb + 64 < CN)
                __builtin_prefetch(&vp[(size_t)(jb + 64 + j) * CDH + d8], 0, 1);
        }
        wait_async_zero();
        __syncthreads();

        // ---- S tiles: four 16x16 j-tiles, K-dim = d = 64 (2 WMMA each) ----
        v8f S[4];
        #pragma unroll
        for (int jt = 0; jt < 4; ++jt) {
            const _Float16* qr = &Qs[(jt * 16 + l15) * 64];
            v16h b0 = load_frag16(qr + bkoff,      qr + bkoff + 16);
            v16h b1 = load_frag16(qr + 32 + bkoff, qr + 32 + bkoff + 16);
            v8f s = {};
            s = __builtin_amdgcn_wmma_f32_16x16x32_f16(false, aK0, false, b0, (short)0, s, false, false);
            s = __builtin_amdgcn_wmma_f32_16x16x32_f16(false, aK1, false, b1, (short)0, s, false, false);
            S[jt] = s;
        }

        // ---- online softmax over the 64 new columns ----
        float alpha[8];
        #pragma unroll
        for (int r = 0; r < 8; ++r) {
            float sv[4];
            #pragma unroll
            for (int jt = 0; jt < 4; ++jt) sv[jt] = S[jt][r] * scale;
            float mx = fmaxf(fmaxf(sv[0], sv[1]), fmaxf(sv[2], sv[3]));
            #pragma unroll
            for (int off = 8; off >= 1; off >>= 1)
                mx = fmaxf(mx, __shfl_xor(mx, off, 32));   // stays within 16-lane half
            float nm = fmaxf(mrow[r], mx);
            float al = __expf(mrow[r] - nm);
            float pv[4], rs = 0.f;
            #pragma unroll
            for (int jt = 0; jt < 4; ++jt) { pv[jt] = __expf(sv[jt] - nm); rs += pv[jt]; }
            #pragma unroll
            for (int off = 8; off >= 1; off >>= 1)
                rs += __shfl_xor(rs, off, 32);
            mrow[r] = nm;
            lrow[r] = lrow[r] * al + rs;
            alpha[r] = al;
            // spill P (C-layout) to the per-wave LDS slab; same-wave LDS ops
            // are in-order (DScnt), so no block barrier needed before reload.
            int rowi = r + ((lane >> 4) ? 8 : 0);
            _Float16* pr = &Ps[wave * 16 * 64 + rowi * 64 + l15];
            #pragma unroll
            for (int jt = 0; jt < 4; ++jt) pr[jt * 16] = (_Float16)pv[jt];
        }
        #pragma unroll
        for (int t = 0; t < 4; ++t)
            #pragma unroll
            for (int r = 0; r < 8; ++r)
                Yacc[t][r] *= alpha[r];

        // ---- Y += P @ V : A = P (16x64, 2 K-chunks), B = V (4 d-tiles) ----
        const _Float16* prow = &Ps[wave * 16 * 64 + l15 * 64];
        v16h aP0 = load_frag16(prow + akb,      prow + 16 + akb);
        v16h aP1 = load_frag16(prow + 32 + akb, prow + 48 + akb);
        #pragma unroll
        for (int dt = 0; dt < 4; ++dt) {
            const _Float16* vr = &Vt[(dt * 16 + l15) * 64];
            v16h bv0 = load_frag16(vr + bkoff,      vr + bkoff + 16);
            v16h bv1 = load_frag16(vr + 32 + bkoff, vr + 32 + bkoff + 16);
            Yacc[dt] = __builtin_amdgcn_wmma_f32_16x16x32_f16(
                false, aP0, false, bv0, (short)0, Yacc[dt], false, false);
            Yacc[dt] = __builtin_amdgcn_wmma_f32_16x16x32_f16(
                false, aP1, false, bv1, (short)0, Yacc[dt], false, false);
        }
        __syncthreads();   // protect Qs/Vt before next staging pass
    }

    // ---- normalize and write combined y [b, n, h*64 + d] as f16 ----
    const int b = bh / CH;
    const int h = bh % CH;
    #pragma unroll
    for (int r = 0; r < 8; ++r) {
        int i = iBase + r + ((lane >> 4) ? 8 : 0);
        float inv = 1.0f / lrow[r];
        size_t base = ((size_t)b * CN + i) * (CH * CDH) + h * CDH;
        #pragma unroll
        for (int dt = 0; dt < 4; ++dt)
            y[base + dt * 16 + l15] = (_Float16)(Yacc[dt][r] * inv);
    }
}

// ---------------------------------------------------------------------------
// Host launcher.
// ---------------------------------------------------------------------------
extern "C" void kernel_launch(void* const* d_in, const int* in_sizes, int n_in,
                              void* d_out, int out_size, void* d_ws, size_t ws_size,
                              hipStream_t stream) {
    const float* x     = (const float*)d_in[0];
    const float* w_qkv = (const float*)d_in[1];
    const float* b_qkv = (const float*)d_in[2];
    const float* w_out = (const float*)d_in[3];
    const float* b_out = (const float*)d_in[4];
    float* out = (float*)d_out;

    const size_t per = (size_t)CB * CH * CN * CDH;   // elements per q/k/v tensor
    _Float16* qkv = (_Float16*)d_ws;                 // [q | k | v | y] f16
    _Float16* qp  = qkv;
    _Float16* kp  = qkv + per;
    _Float16* vp  = qkv + 2 * per;
    _Float16* yp  = qkv + 3 * per;                   // [B*N, 768] f16

    dim3 blk(256);
    const int M = CB * CN;                           // 8192

    // 1) QKV projection: [8192,768] @ [768,2304] -> scatter q/k/v f16
    mhsa_gemm_kernel<0><<<dim3(CDQKV / 128, M / 128), blk, 0, stream>>>(
        x, w_qkv, b_qkv, (void*)qkv, M, CDIN, CDQKV);

    // 2) Attention (S = K·Qᵀ, softmax over q index, Y = P·V)
    mhsa_attn_kernel<<<dim3(CB * CH * (CN / 128)), blk, 0, stream>>>(qp, kp, vp, yp);

    // 3) Output projection: [8192,768] @ [768,768] + bias -> f32 out
    mhsa_gemm_kernel<1><<<dim3(CDOUT / 128, M / 128), blk, 0, stream>>>(
        yp, w_out, b_out, (void*)out, M, CDIN, CDOUT);
}
